// MoleculeTorsionDenoiser_37134287242021
// MI455X (gfx1250) — matrix operand order repaired and checked by reading, not compiled, not run on hardware
//
#include <hip/hip_runtime.h>

typedef _Float16 h16;
typedef __attribute__((ext_vector_type(16))) _Float16 v16h;
typedef __attribute__((ext_vector_type(8)))  _Float16 v8h;
typedef __attribute__((ext_vector_type(8)))  float    v8f;

#define NS   32
#define NV   8
#define FEATD 112
#define HE   128
#define WN   3328
#define NA   6000
#define NE   48000

// ---- per-path tables (order: ss, vvs, sv, vs, pxv, vxp, pp, qv, pq, qq) ----
// shapes (I,O): (32,32)(8,32)(32,8)(8,8)(8,8)(8,8)(8,8)(32,8)(8,32)(32,32)
__constant__ int c_off[10]  = {0,1024,1280,1536,1600,1664,1728,1792,2048,2304};
__constant__ int c_I[10]    = {32,8,32,8,8,8,8,32,8,32};
__constant__ int c_Osh[10]  = {5,5,3,3,3,3,3,3,5,5};
// per-edge input block layout (floats): ssh0@0(32) vdot@32(8) pdot@40(8) qsh0@48(32)
// ssh1@80(96) vsh0@176(24) pcross@200(24) vcross@224(24) psh0@248(24) qsh1@272(96) => 368
__constant__ int c_inb[10]  = {0,32,80,176,200,224,248,272,40,48};
__constant__ int c_outb[10] = {0,0,32,32,32,56,56,56,80,80};
__constant__ int c_vec[10]  = {0,0,1,1,1,1,1,1,0,0};

// ------------------------------ utility kernels ------------------------------
__global__ void zero_f32_kernel(float* p, int n) {
  int i = blockIdx.x * 256 + threadIdx.x;
  if (i < n) p[i] = 0.f;
}

__global__ void cvt_f16_kernel(const float* __restrict__ src, h16* __restrict__ dst, int n) {
  int i = blockIdx.x * 256 + threadIdx.x;
  if (i < n) dst[i] = (h16)src[i];
}

// w1t[(l*128+n)*128+k] = w1[(l*128+k)*128+n]
__global__ void prep_w1t_kernel(const float* __restrict__ w1, h16* __restrict__ w1t) {
  int idx = blockIdx.x * 256 + threadIdx.x;
  if (idx < 2 * 128 * 128) {
    int l = idx / (128 * 128);
    int rem = idx - l * 128 * 128;
    int n = rem / 128;
    int k = rem - n * 128;
    w1t[idx] = (h16)w1[(size_t)(l * 128 + k) * 128 + n];
  }
}

// w2t[(l*WN+n)*128+k] = w2[(l*128+k)*WN+n]
__global__ void prep_w2t_kernel(const float* __restrict__ w2, h16* __restrict__ w2t) {
  int idx = blockIdx.x * 256 + threadIdx.x;
  if (idx < 2 * WN * 128) {
    int l = idx / (WN * 128);
    int rem = idx - l * WN * 128;
    int n = rem / 128;
    int k = rem - n * 128;
    w2t[idx] = (h16)w2[((size_t)(l * 128 + k)) * WN + n];
  }
}

__global__ void count_kernel(const int* __restrict__ eidx, float* __restrict__ cnt) {
  int e = blockIdx.x * 256 + threadIdx.x;
  if (e < NE) atomicAdd(&cnt[eidx[NE + e]], 1.f);
}

// ------------------------------ edge MLP stage 1 -----------------------------
// h = relu(edge_feat @ w1 + b1), 16 edges x 128 outs per workgroup (4 waves)
__global__ __launch_bounds__(128)
void edge_mlp_kernel(const h16* __restrict__ efh,
                     const h16* __restrict__ w1t,   // this layer, [n][k] f16
                     const float* __restrict__ b1,  // this layer
                     h16* __restrict__ hout) {
  const int tid   = threadIdx.x;
  const int e0    = blockIdx.x * 16;
  const int lane  = tid & 31;
  const int wave  = tid >> 5;
  const int halfw = (lane < 16) ? 0 : 1;
  const int col16 = lane & 15;

  v16h A[4];
  {
    const h16* ap = efh + (size_t)(e0 + col16) * HE + (halfw ? 8 : 0);
#pragma unroll
    for (int s = 0; s < 4; ++s) {
      v8h lo = *(const v8h*)(ap + 32 * s);
      v8h hi = *(const v8h*)(ap + 32 * s + 16);
      A[s] = __builtin_shufflevector(lo, hi, 0,1,2,3,4,5,6,7,8,9,10,11,12,13,14,15);
    }
  }
#pragma unroll
  for (int u = 0; u < 2; ++u) {
    int nt   = 2 * wave + u;
    int colg = nt * 16 + col16;
    v8f acc = {};
    const h16* bp = w1t + (size_t)colg * HE + (halfw ? 8 : 0);
#pragma unroll
    for (int s = 0; s < 4; ++s) {
      v8h blo = *(const v8h*)(bp + 32 * s);
      v8h bhi = *(const v8h*)(bp + 32 * s + 16);
      v16h B = __builtin_shufflevector(blo, bhi, 0,1,2,3,4,5,6,7,8,9,10,11,12,13,14,15);
      acc = __builtin_amdgcn_wmma_f32_16x16x32_f16(false, A[s], false, B, (short)0, acc, false, false);
    }
    float bias = b1[colg];
#pragma unroll
    for (int j = 0; j < 8; ++j) {
      int r = j + (halfw ? 8 : 0);
      float v = acc[j] + bias;
      v = v > 0.f ? v : 0.f;
      hout[(size_t)(e0 + r) * HE + colg] = (h16)v;
    }
  }
}

// --------------- fused: (h @ w2 + b2) -> equivariant contraction -------------
// 32 edges per workgroup: two 16-row A tiles share every B fragment (halves
// L2 traffic for w2, which is the roofline-limiting stream).
__global__ __launch_bounds__(128)
void fused_conv_kernel(const float* __restrict__ x,
                       const h16* __restrict__ h,
                       const h16* __restrict__ w2t,  // this layer, [n][k] f16
                       const float* __restrict__ b2, // this layer
                       const int* __restrict__ eidx, // [2][NE]
                       const float* __restrict__ sh, // [NE][4]
                       float* __restrict__ sums) {   // [NA][FEATD]
  __shared__ float sin_[32 * 368];   // 47104 B
  __shared__ float smsg[32 * FEATD]; // 14336 B
  const int tid = threadIdx.x;
  const int e0  = blockIdx.x * 32;

  // ---- stage per-edge equivariant inputs (4 threads / edge) ----
  {
    int r = tid >> 2, l4 = tid & 3;
    int e = e0 + r;
    int dst = eidx[e];
    const float* xe = x + (size_t)dst * FEATD;
    float s0  = sh[e * 4 + 0];
    float s1x = sh[e * 4 + 1], s1y = sh[e * 4 + 2], s1z = sh[e * 4 + 3];
    float* ib = sin_ + r * 368;
#pragma unroll
    for (int t = 0; t < 8; ++t) {
      int i = l4 * 8 + t;
      float sv = xe[i];
      ib[0 + i]          = sv * s0;
      ib[80 + 3 * i + 0] = sv * s1x;
      ib[80 + 3 * i + 1] = sv * s1y;
      ib[80 + 3 * i + 2] = sv * s1z;
      float qv = xe[80 + i];
      ib[48 + i]          = qv * s0;
      ib[272 + 3 * i + 0] = qv * s1x;
      ib[272 + 3 * i + 1] = qv * s1y;
      ib[272 + 3 * i + 2] = qv * s1z;
    }
#pragma unroll
    for (int u = 0; u < 2; ++u) {
      int i = l4 * 2 + u;
      float vx = xe[32 + 3 * i + 0], vy = xe[32 + 3 * i + 1], vz = xe[32 + 3 * i + 2];
      ib[32 + i] = vx * s1x + vy * s1y + vz * s1z;          // v . sh1
      ib[176 + 3 * i + 0] = vx * s0;
      ib[176 + 3 * i + 1] = vy * s0;
      ib[176 + 3 * i + 2] = vz * s0;
      ib[224 + 3 * i + 0] = vy * s1z - vz * s1y;            // v x sh1
      ib[224 + 3 * i + 1] = vz * s1x - vx * s1z;
      ib[224 + 3 * i + 2] = vx * s1y - vy * s1x;
      float px = xe[56 + 3 * i + 0], py = xe[56 + 3 * i + 1], pz = xe[56 + 3 * i + 2];
      ib[40 + i] = px * s1x + py * s1y + pz * s1z;          // p . sh1
      ib[248 + 3 * i + 0] = px * s0;
      ib[248 + 3 * i + 1] = py * s0;
      ib[248 + 3 * i + 2] = pz * s0;
      ib[200 + 3 * i + 0] = py * s1z - pz * s1y;            // p x sh1
      ib[200 + 3 * i + 1] = pz * s1x - px * s1z;
      ib[200 + 3 * i + 2] = px * s1y - py * s1x;
    }
    for (int f = l4; f < FEATD; f += 4) smsg[r * FEATD + f] = 0.f;
  }
  __syncthreads();

  // ---- WMMA + on-the-fly contraction ----
  const int lane  = tid & 31;
  const int wave  = tid >> 5;
  const int halfw = (lane < 16) ? 0 : 1;
  const int col16 = lane & 15;

  v16h A0[4], A1[4];
  {
    const h16* hp0 = h + (size_t)(e0 + col16) * HE + (halfw ? 8 : 0);
    const h16* hp1 = h + (size_t)(e0 + 16 + col16) * HE + (halfw ? 8 : 0);
#pragma unroll
    for (int s = 0; s < 4; ++s) {
      v8h lo0 = *(const v8h*)(hp0 + 32 * s);
      v8h hi0 = *(const v8h*)(hp0 + 32 * s + 16);
      A0[s] = __builtin_shufflevector(lo0, hi0, 0,1,2,3,4,5,6,7,8,9,10,11,12,13,14,15);
      v8h lo1 = *(const v8h*)(hp1 + 32 * s);
      v8h hi1 = *(const v8h*)(hp1 + 32 * s + 16);
      A1[s] = __builtin_shufflevector(lo1, hi1, 0,1,2,3,4,5,6,7,8,9,10,11,12,13,14,15);
    }
  }

  for (int chunk = wave; chunk < 208; chunk += 4) {
    // map chunk -> (path p, local tile c)
    int p = 0, c = chunk;
    while (c >= ((c_I[p] << c_Osh[p]) >> 4)) { c -= (c_I[p] << c_Osh[p]) >> 4; ++p; }
    int fl0  = c << 4;
    int colg = c_off[p] + fl0;

    v8f acc0 = {}, acc1 = {};
    const h16* bp = w2t + (size_t)(colg + col16) * HE + (halfw ? 8 : 0);
#pragma unroll
    for (int s = 0; s < 4; ++s) {
      v8h blo = *(const v8h*)(bp + 32 * s);
      v8h bhi = *(const v8h*)(bp + 32 * s + 16);
      v16h B = __builtin_shufflevector(blo, bhi, 0,1,2,3,4,5,6,7,8,9,10,11,12,13,14,15);
      acc0 = __builtin_amdgcn_wmma_f32_16x16x32_f16(false, A0[s], false, B, (short)0, acc0, false, false);
      acc1 = __builtin_amdgcn_wmma_f32_16x16x32_f16(false, A1[s], false, B, (short)0, acc1, false, false);
    }
    float bias = b2[colg + col16];
    int osh   = c_Osh[p];
    int omask = (1 << osh) - 1;
    int inb = c_inb[p], outb = c_outb[p], vec = c_vec[p];
    int f_l  = fl0 + col16;
    int i_in = f_l >> osh;
    int o    = f_l & omask;
#pragma unroll
    for (int j = 0; j < 8; ++j) {
      int rbase = j + (halfw ? 8 : 0);
#pragma unroll
      for (int tile = 0; tile < 2; ++tile) {
        int r = rbase + tile * 16;
        float wv = (tile ? acc1[j] : acc0[j]) + bias;
        const float* ib = sin_ + r * 368 + inb;
        float* mg = smsg + r * FEATD + outb;
        if (vec) {
          atomicAdd(&mg[3 * o + 0], ib[3 * i_in + 0] * wv);
          atomicAdd(&mg[3 * o + 1], ib[3 * i_in + 1] * wv);
          atomicAdd(&mg[3 * o + 2], ib[3 * i_in + 2] * wv);
        } else {
          atomicAdd(&mg[o], ib[i_in] * wv);
        }
      }
    }
  }
  __syncthreads();

  // ---- scatter to per-atom sums ----
  {
    int r = tid >> 2, l4 = tid & 3;
    int e = e0 + r;
    int src = eidx[NE + e];
    float* sp = sums + (size_t)src * FEATD;
    for (int f = l4; f < FEATD; f += 4) atomicAdd(&sp[f], smsg[r * FEATD + f]);
  }
}

// ------------------------------ epilogue kernels -----------------------------
__global__ void add_mean_kernel(const float* __restrict__ x, const float* __restrict__ sums,
                                const float* __restrict__ cnt, float* __restrict__ y) {
  int idx = blockIdx.x * 256 + threadIdx.x;
  if (idx < NA * FEATD) {
    int a = idx / FEATD;
    float cv = cnt[a];
    cv = cv > 1.f ? cv : 1.f;
    y[idx] = x[idx] + sums[idx] / cv;
  }
}

__global__ __launch_bounds__(256)
void stats_kernel(const float* __restrict__ y, float* __restrict__ meanS,
                  float* __restrict__ scaleS, float* __restrict__ scaleV) {
  __shared__ float r1[256], r2[256];
  int b = blockIdx.x, t = threadIdx.x;
  float s1 = 0.f, s2 = 0.f;
  if (b < 64) {
    int col = (b < 32) ? b : b + 48;   // s cols 0..31, q cols 80..111
    for (int a = t; a < NA; a += 256) {
      float v = y[(size_t)a * FEATD + col];
      s1 += v; s2 += v * v;
    }
  } else {
    int ch  = b - 64;
    int col = (ch < 8) ? (32 + 3 * ch) : (56 + 3 * (ch - 8));
    for (int a = t; a < NA; a += 256) {
      float v0 = y[(size_t)a * FEATD + col + 0];
      float v1 = y[(size_t)a * FEATD + col + 1];
      float v2 = y[(size_t)a * FEATD + col + 2];
      s1 += v0 * v0 + v1 * v1 + v2 * v2;
    }
  }
  r1[t] = s1; r2[t] = s2;
  __syncthreads();
  for (int off = 128; off > 0; off >>= 1) {
    if (t < off) { r1[t] += r1[t + off]; r2[t] += r2[t + off]; }
    __syncthreads();
  }
  if (t == 0) {
    const float eps = 1e-5f;
    if (b < 64) {
      float m   = r1[0] / (float)NA;
      float var = r2[0] / (float)NA - m * m;
      meanS[b]  = m;
      scaleS[b] = rsqrtf(var + eps);
    } else {
      scaleV[b - 64] = rsqrtf(r1[0] / (float)NA + eps);
    }
  }
}

__global__ void norm_kernel(const float* __restrict__ y, const float* __restrict__ meanS,
                            const float* __restrict__ scaleS, const float* __restrict__ scaleV,
                            float* __restrict__ xo) {
  int idx = blockIdx.x * 256 + threadIdx.x;
  if (idx < NA * FEATD) {
    int f = idx % FEATD;
    float v = y[idx], out;
    if (f < 32)       out = (v - meanS[f]) * scaleS[f];
    else if (f < 56)  out = v * scaleV[(f - 32) / 3];
    else if (f < 80)  out = v * scaleV[8 + (f - 56) / 3];
    else { int j = f - 48; out = (v - meanS[j]) * scaleS[j]; }
    xo[idx] = out;
  }
}

// --------------------------------- launcher ----------------------------------
extern "C" void kernel_launch(void* const* d_in, const int* in_sizes, int n_in,
                              void* d_out, int out_size, void* d_ws, size_t ws_size,
                              hipStream_t stream) {
  (void)in_sizes; (void)n_in; (void)out_size; (void)ws_size;
  const float* atom = (const float*)d_in[0];
  const float* ef   = (const float*)d_in[1];
  const float* esh  = (const float*)d_in[2];
  const int*   eidx = (const int*)d_in[3];
  const float* w1   = (const float*)d_in[4];
  const float* b1   = (const float*)d_in[5];
  const float* w2   = (const float*)d_in[6];
  const float* b2   = (const float*)d_in[7];
  float* out = (float*)d_out;

  char* ws = (char*)d_ws;
  size_t off = 0;
  auto alloc = [&](size_t bytes) -> void* {
    void* p = ws + off;
    off = (off + bytes + 255) & ~(size_t)255;
    return p;
  };
  h16*   efh    = (h16*)alloc((size_t)NE * HE * 2);
  h16*   hbuf   = (h16*)alloc((size_t)NE * HE * 2);
  h16*   w1t    = (h16*)alloc((size_t)2 * 128 * 128 * 2);
  h16*   w2t    = (h16*)alloc((size_t)2 * WN * 128 * 2);
  float* sums   = (float*)alloc((size_t)NA * FEATD * 4);
  float* cnt    = (float*)alloc((size_t)NA * 4);
  float* ybuf   = (float*)alloc((size_t)NA * FEATD * 4);
  float* xbuf   = (float*)alloc((size_t)NA * FEATD * 4);
  float* meanS  = (float*)alloc(64 * 4);
  float* scaleS = (float*)alloc(64 * 4);
  float* scaleV = (float*)alloc(16 * 4);

  // prep (deterministic, re-run every call)
  cvt_f16_kernel<<<(NE * HE + 255) / 256, 256, 0, stream>>>(ef, efh, NE * HE);
  prep_w1t_kernel<<<(2 * 128 * 128 + 255) / 256, 256, 0, stream>>>(w1, w1t);
  prep_w2t_kernel<<<(2 * WN * 128 + 255) / 256, 256, 0, stream>>>(w2, w2t);
  zero_f32_kernel<<<(NA + 255) / 256, 256, 0, stream>>>(cnt, NA);
  count_kernel<<<(NE + 255) / 256, 256, 0, stream>>>(eidx, cnt);

  for (int l = 0; l < 2; ++l) {
    const float* xin = (l == 0) ? atom : xbuf;
    float* xout = (l == 1) ? out : xbuf;
    zero_f32_kernel<<<(NA * FEATD + 255) / 256, 256, 0, stream>>>(sums, NA * FEATD);
    edge_mlp_kernel<<<NE / 16, 128, 0, stream>>>(efh, w1t + (size_t)l * 128 * 128,
                                                 b1 + (size_t)l * 128, hbuf);
    fused_conv_kernel<<<NE / 32, 128, 0, stream>>>(xin, hbuf, w2t + (size_t)l * WN * 128,
                                                   b2 + (size_t)l * WN, eidx, esh, sums);
    add_mean_kernel<<<(NA * FEATD + 255) / 256, 256, 0, stream>>>(xin, sums, cnt, ybuf);
    stats_kernel<<<80, 256, 0, stream>>>(ybuf, meanS, scaleS, scaleV);
    norm_kernel<<<(NA * FEATD + 255) / 256, 256, 0, stream>>>(ybuf, meanS, scaleS, scaleV, xout);
  }
}
